// KGLearner_5746666242440
// MI455X (gfx1250) — matrix operand order repaired and checked by reading, not compiled, not run on hardware
//
#include <hip/hip_runtime.h>

#define NUM_EVT   8192
#define NUM_SEVT  16384
#define EMBED     128
#define KCHUNKS   4
#define KCHUNK    (NUM_SEVT / KCHUNKS)   // 4096
#define KSTEP     32
#define NSTEPS    (KCHUNK / KSTEP)       // 128
#define ROWTILES  (NUM_EVT / 16)         // 512
#define L2E       1.44269504088896340736f

typedef __attribute__((ext_vector_type(16))) _Float16 v16h;
typedef __attribute__((ext_vector_type(8)))  float    v8f;
typedef __attribute__((ext_vector_type(4)))  float    f4;

#if defined(__has_builtin)
#if __has_builtin(__builtin_amdgcn_global_load_async_to_lds_b128) && \
    __has_builtin(__builtin_amdgcn_s_wait_asynccnt)
#define USE_ASYNC_LDS 1
#endif
#endif
#ifndef USE_ASYNC_LDS
#define USE_ASYNC_LDS 0
#endif

#if USE_ASYNC_LDS
// Builtin expects pointers to 16-byte int vectors (per hipcc diagnostic),
// in global (AS1) / LDS (AS3) address spaces.
typedef int v4i_vs __attribute__((vector_size(16)));
typedef __attribute__((address_space(1))) v4i_vs gv4i;
typedef __attribute__((address_space(3))) v4i_vs lv4i;
#endif

// ---------------------------------------------------------------------------
// Prep 1: s_proj (scaled by log2 e) and subevent transposed -> f16 [128][16384]
// ---------------------------------------------------------------------------
__global__ void prep_subt_sproj(const float* __restrict__ subevent,
                                const float* __restrict__ attn_w,
                                _Float16* __restrict__ sub_t,
                                float* __restrict__ s2) {
  __shared__ float red[EMBED];
  const int k = blockIdx.x;            // 0..16383
  const int n = threadIdx.x;           // 0..127
  float v = subevent[k * EMBED + n];
  sub_t[(size_t)n * NUM_SEVT + k] = (_Float16)v;
  red[n] = v * attn_w[n];
  __syncthreads();
  for (int s = EMBED / 2; s > 0; s >>= 1) {
    if (n < s) red[n] += red[n + s];
    __syncthreads();
  }
  if (n == 0) s2[k] = red[0] * L2E;
}

// ---------------------------------------------------------------------------
// Prep 2: global max of s2 (single workgroup)
// ---------------------------------------------------------------------------
__global__ void reduce_smax(const float* __restrict__ s2,
                            float* __restrict__ smax2) {
  __shared__ float red[256];
  float m = -1e30f;
  for (int k = threadIdx.x; k < NUM_SEVT; k += 256) m = fmaxf(m, s2[k]);
  red[threadIdx.x] = m;
  __syncthreads();
  for (int s = 128; s > 0; s >>= 1) {
    if (threadIdx.x < s) red[threadIdx.x] = fmaxf(red[threadIdx.x], red[threadIdx.x + s]);
    __syncthreads();
  }
  if (threadIdx.x == 0) *smax2 = red[0];
}

// ---------------------------------------------------------------------------
// Prep 3: e_proj (scaled) and per-row safe softmax shift bound
// ---------------------------------------------------------------------------
__global__ void prep_eproj(const float* __restrict__ event,
                           const float* __restrict__ attn_w,
                           const float* __restrict__ smax2,
                           float* __restrict__ e2,
                           float* __restrict__ bound2) {
  __shared__ float red[EMBED];
  const int i = blockIdx.x;
  const int n = threadIdx.x;
  red[n] = event[i * EMBED + n] * attn_w[EMBED + n];
  __syncthreads();
  for (int s = EMBED / 2; s > 0; s >>= 1) {
    if (n < s) red[n] += red[n + s];
    __syncthreads();
  }
  if (n == 0) {
    float e = red[0] * L2E;
    e2[i] = e;
    bound2[i] = fmaxf(0.0f, e + *smax2);   // score*log2e <= bound2 always
  }
}

// ---------------------------------------------------------------------------
// B-tile fill: 32x128 f16 tile in WMMA B fragment layout (seg = tile*32+lane).
// Async global->LDS on CDNA5 (ASYNCcnt), else register-staged b128 copies.
// ---------------------------------------------------------------------------
__device__ __forceinline__ void fill_tile(const _Float16* __restrict__ sub_t,
                                          uint4* __restrict__ buf,
                                          int k0, int tid) {
#pragma unroll
  for (int q = 0; q < 2; ++q) {
    const int s  = tid * 2 + q;
    const int bt = s >> 5;
    const int bl = s & 31;
    const _Float16* gp = sub_t +
        (size_t)(bt * 16 + (bl & 15)) * NUM_SEVT + (k0 + (bl >> 4) * 8);
#if USE_ASYNC_LDS
    __builtin_amdgcn_global_load_async_to_lds_b128(
        (gv4i*)gp,       (lv4i*)&buf[2 * s],     0, 0);
    __builtin_amdgcn_global_load_async_to_lds_b128(
        (gv4i*)(gp + 16), (lv4i*)&buf[2 * s + 1], 0, 0);
#else
    const uint4* g4 = (const uint4*)gp;
    buf[2 * s]     = g4[0];
    buf[2 * s + 1] = g4[2];
#endif
  }
}

__device__ __forceinline__ void drain_async_then_barrier() {
#if USE_ASYNC_LDS
  __builtin_amdgcn_s_wait_asynccnt(0);
#endif
  __syncthreads();
}

__device__ __forceinline__ void load_adj4(f4 (&aa)[4],
                                          const float* __restrict__ adjrow,
                                          int offA) {
  aa[0] = __builtin_nontemporal_load((const f4*)(adjrow + offA));
  aa[1] = __builtin_nontemporal_load((const f4*)(adjrow + offA + 4));
  aa[2] = __builtin_nontemporal_load((const f4*)(adjrow + offA + 16));
  aa[3] = __builtin_nontemporal_load((const f4*)(adjrow + offA + 20));
}

// ---------------------------------------------------------------------------
// Fused streaming kernel: scores -> exp weights -> WMMA matmul.
// WG = 4 waves; wave w owns row-tile rt = (blockIdx>>2)*4 + w (16 rows).
// blockIdx&3 selects the K chunk of 4096. One NT pass over adj.
// Double-buffered B tile: one barrier per K step; adj software-pipelined.
// ---------------------------------------------------------------------------
__launch_bounds__(128, 1)
__global__ void fused_attn_wmma(const float* __restrict__ adj,
                                const _Float16* __restrict__ sub_t,
                                const float* __restrict__ s2,
                                const float* __restrict__ e2,
                                const float* __restrict__ bound2,
                                float* __restrict__ accP,
                                float* __restrict__ lP) {
  __shared__ float s2lds[KCHUNK];      // 16 KB: s_proj*log2e for this K chunk
  __shared__ uint4 bTile[2][512];      // 2 x 8 KB: double-buffered B tiles

  const int tid  = threadIdx.x;
  const int lane = tid & 31;
  const int wv   = tid >> 5;
  const int wg   = blockIdx.x;
  const int g    = wg >> 2;           // row group (4 tiles of 16 rows)
  const int c    = wg & 3;            // K chunk
  const int rt   = g * 4 + wv;        // row tile 0..511
  const int half = lane >> 4;
  const int lrow = lane & 15;
  const int row  = rt * 16 + lrow;
  const int kc0  = c * KCHUNK;

  // Stage s2 chunk into LDS (4096 floats, 128 threads)
  for (int j = tid * 4; j < KCHUNK; j += 128 * 4) {
    *(f4*)&s2lds[j] = *(const f4*)&s2[kc0 + j];
  }

  const float e2v = e2[row];
  const float b2v = bound2[row];
  const float* adjrow = adj + (size_t)row * NUM_SEVT;

  v8f acc[8];
#pragma unroll
  for (int t = 0; t < 8; ++t) { v8f z = {}; acc[t] = z; }
  f4 lsum4 = {};

  // Prologue: fill buffer 0 with tile 0; prefetch adj for step 0.
  fill_tile(sub_t, bTile[0], kc0, tid);
  f4 aa_cur[4];
  load_adj4(aa_cur, adjrow, kc0 + half * 8);
  drain_async_then_barrier();          // tile 0 + s2lds visible

  for (int st = 0; st < NSTEPS; ++st) {
    const int k0 = kc0 + st * KSTEP;

    // Issue next tile's B fill (into the other buffer) and adj prefetch.
    f4 aa_nxt[4];
    if (st + 1 < NSTEPS) {
      fill_tile(sub_t, bTile[(st + 1) & 1], k0 + KSTEP, tid);
      load_adj4(aa_nxt, adjrow, k0 + KSTEP + half * 8);
    }

    // --- A fragment: 16 scores per lane in 16-bit 16x32 layout ---
    const int offL = k0 - kc0 + half * 8;
    f4 sp[4];
    sp[0] = *(const f4*)&s2lds[offL];
    sp[1] = *(const f4*)&s2lds[offL + 4];
    sp[2] = *(const f4*)&s2lds[offL + 16];
    sp[3] = *(const f4*)&s2lds[offL + 20];

    v16h afrag;
#pragma unroll
    for (int u = 0; u < 16; ++u) {
      float a  = aa_cur[u >> 2][u & 3];
      float t2 = sp[u >> 2][u & 3] + e2v;          // (e_proj+s_proj)*log2e
      float x  = a * t2;                           // adj*score, log2 domain
      float sc = (x > 0.0f ? x : 0.2f * x) - b2v;  // leaky_relu - bound (<=0)
      float p  = __builtin_amdgcn_exp2f(sc);       // unnormalized attention
      lsum4[u & 3] += p;                           // 4-wide partial sums
      afrag[u] = (_Float16)(p * a);                // weight = attn_unnorm*adj
    }

    // --- 8 WMMAs: acc(16x128) += A(16x32) @ B(32x128) ---
    const uint4* bb = bTile[st & 1];
#pragma unroll
    for (int t = 0; t < 8; ++t) {
      union { uint4 q[2]; v16h h; } bf;
      bf.q[0] = bb[(t * 32 + lane) * 2];
      bf.q[1] = bb[(t * 32 + lane) * 2 + 1];
      acc[t] = __builtin_amdgcn_wmma_f32_16x16x32_f16(
          false, afrag, false, bf.h, (short)0, acc[t], false, false);
    }

    drain_async_then_barrier();        // next tile resident; prev buffer free
#pragma unroll
    for (int u = 0; u < 4; ++u) aa_cur[u] = aa_nxt[u];
  }

  // --- epilogue: combine lane halves of lsum, store partials ---
  float lsum = (lsum4[0] + lsum4[1]) + (lsum4[2] + lsum4[3]);
  lsum += __shfl_xor(lsum, 16, 32);
  if (lane < 16) lP[(rt * 4 + c) * 16 + lane] = lsum;

  float* ap = accP + (size_t)(rt * 4 + c) * 2048;  // 16x128 per (rt,chunk)
#pragma unroll
  for (int t = 0; t < 8; ++t) {
    union { v8f v; f4 f2[2]; } ua;
    ua.v = acc[t];
    float* dst = ap + (t * 32 + lane) * 8;
    *(f4*)dst       = ua.f2[0];
    *(f4*)(dst + 4) = ua.f2[1];
  }
}

// ---------------------------------------------------------------------------
// Combine K-chunk partials (shared per-row shift cancels) and finish.
// ---------------------------------------------------------------------------
__global__ void combine_out(const float* __restrict__ event,
                            const float* __restrict__ accP,
                            const float* __restrict__ lP,
                            float* __restrict__ out) {
  const int idx = blockIdx.x * 256 + threadIdx.x;   // < 8192*128
  const int i = idx >> 7, n = idx & 127;
  const int rt = i >> 4, m = i & 15;
  const int t = n >> 4;
  const int ln = (n & 15) + ((m >> 3) << 4);
  const int r = m & 7;
  float ls = 0.0f, ac = 0.0f;
#pragma unroll
  for (int cc = 0; cc < 4; ++cc) {
    ls += lP[(rt * 4 + cc) * 16 + m];
    ac += accP[(size_t)(rt * 4 + cc) * 2048 + (t * 32 + ln) * 8 + r];
  }
  out[idx] = 0.5f * (event[idx] + ac / ls);
}

// ---------------------------------------------------------------------------
extern "C" void kernel_launch(void* const* d_in, const int* in_sizes, int n_in,
                              void* d_out, int out_size, void* d_ws, size_t ws_size,
                              hipStream_t stream) {
  const float* adj      = (const float*)d_in[0];
  const float* subevent = (const float*)d_in[1];
  const float* event    = (const float*)d_in[2];
  const float* attn_w   = (const float*)d_in[3];
  float* out = (float*)d_out;

  char* ws = (char*)d_ws;
  size_t off = 0;
  _Float16* sub_t = (_Float16*)(ws + off); off += (size_t)NUM_SEVT * EMBED * 2; // 4 MB
  float* s2     = (float*)(ws + off); off += (size_t)NUM_SEVT * 4;              // 64 KB
  float* e2     = (float*)(ws + off); off += (size_t)NUM_EVT * 4;               // 32 KB
  float* bound2 = (float*)(ws + off); off += (size_t)NUM_EVT * 4;               // 32 KB
  float* smax2  = (float*)(ws + off); off += 256;
  float* lP     = (float*)(ws + off); off += (size_t)ROWTILES * KCHUNKS * 16 * 4;        // 128 KB
  float* accP   = (float*)(ws + off); off += (size_t)ROWTILES * KCHUNKS * 16 * EMBED * 4; // 16 MB
  (void)off; (void)ws_size; (void)in_sizes; (void)n_in; (void)out_size;

  prep_subt_sproj<<<NUM_SEVT, EMBED, 0, stream>>>(subevent, attn_w, sub_t, s2);
  reduce_smax<<<1, 256, 0, stream>>>(s2, smax2);
  prep_eproj<<<NUM_EVT, EMBED, 0, stream>>>(event, attn_w, smax2, e2, bound2);
  fused_attn_wmma<<<(ROWTILES / 4) * KCHUNKS, 128, 0, stream>>>(
      adj, sub_t, s2, e2, bound2, accP, lP);
  combine_out<<<(NUM_EVT * EMBED) / 256, 256, 0, stream>>>(event, accP, lP, out);
}
